// Informer_41506563949111
// MI455X (gfx1250) — compile-verified, matrix operand order checked by reading, not run on hardware
//
#include <hip/hip_runtime.h>
#include <math.h>

// ---------------- model dims ----------------
#define BATCH   32
#define LENC0   384
#define LENC1   192
#define LDEC    288
#define DM      512
#define DFF     2048
#define CIN     510
#define NH      8
#define DKH     64
#define K3      1536    // padded im2col depth (3 blocks of 512)

// ---------------- WMMA GEMM ----------------
typedef _Float16 v16h __attribute__((ext_vector_type(16)));
typedef float    v8f  __attribute__((ext_vector_type(8)));

#define TM 128
#define TN 128
#define TK 32
#define LSTR 40     // 32 + 8 halves pad; 80B row stride -> conflict-free b128 reads

// C[M,N] = act(A[M,K] @ B[K,N] + bias) + res   (row-major fp32, f16 MACs)
// REQUIRES: M % 128 == 0, N % 128 == 0, K % 32 == 0  (guaranteed by padding)
// Software-pipelined: next K-tile global loads overlap WMMA compute.
__global__ void __launch_bounds__(256)
gemm_f16_wmma(const float* __restrict__ A, const float* __restrict__ Bm,
              const float* __restrict__ bias, const float* __restrict__ res,
              float* __restrict__ C, int M, int N, int K, int act)
{
    __shared__ __align__(16) _Float16 As [TM * LSTR];  // [m][k]
    __shared__ __align__(16) _Float16 Bst[TN * LSTR];  // [n][k]  (transposed)

    const int tid  = threadIdx.x;
    const int lane = tid & 31;
    const int wid  = tid >> 5;      // 0..7
    const int wm   = wid & 3;       // 4 row groups of 32
    const int wn   = wid >> 2;      // 2 col groups of 64
    const int g    = lane >> 4;     // K-half select
    const int l16  = lane & 15;     // row (A) / col (B,C)

    const int bm = blockIdx.y * TM;
    const int bn = blockIdx.x * TN;

    // per-thread staging coordinates
    const int arr = tid >> 3, ac4 = (tid & 7) << 2;    // A: 128x32, 4 float4/thread
    const int bkp = (tid >> 5) << 1;                   // B: even base k-row (pairs)
    const int bn4 = (tid & 31) << 2;                   // B: 4 cols/thread

    v8f acc[2][4];
#pragma unroll
    for (int mi = 0; mi < 2; ++mi)
#pragma unroll
        for (int ni = 0; ni < 4; ++ni)
#pragma unroll
            for (int r = 0; r < 8; ++r) acc[mi][ni][r] = 0.0f;

    // ---- prologue: issue loads for tile 0 (8 independent b128 loads) ----
    float4 ra[4], rb[4];
#pragma unroll
    for (int i = 0; i < 4; ++i)
        ra[i] = *(const float4*)(A + (size_t)(bm + arr + 32 * i) * K + ac4);
#pragma unroll
    for (int i = 0; i < 2; ++i) {
        rb[2 * i]     = *(const float4*)(Bm + (size_t)(bkp + 16 * i) * N + bn + bn4);
        rb[2 * i + 1] = *(const float4*)(Bm + (size_t)(bkp + 16 * i + 1) * N + bn + bn4);
    }

    for (int k0 = 0; k0 < K; k0 += TK) {
        // ---- drain held registers into LDS (f32 -> f16) ----
#pragma unroll
        for (int i = 0; i < 4; ++i) {
            union { _Float16 h[4]; uint2 u; } p;
            p.h[0] = (_Float16)ra[i].x; p.h[1] = (_Float16)ra[i].y;
            p.h[2] = (_Float16)ra[i].z; p.h[3] = (_Float16)ra[i].w;
            *(uint2*)&As[(arr + 32 * i) * LSTR + ac4] = p.u;
        }
#pragma unroll
        for (int i = 0; i < 2; ++i) {
            int k = bkp + 16 * i;
            const float* lo = (const float*)&rb[2 * i];
            const float* hi = (const float*)&rb[2 * i + 1];
#pragma unroll
            for (int j = 0; j < 4; ++j) {   // pack (k, k+1) pair -> one b32 store
                union { _Float16 h[2]; unsigned u; } p;
                p.h[0] = (_Float16)lo[j];
                p.h[1] = (_Float16)hi[j];
                *(unsigned*)&Bst[(bn4 + j) * LSTR + k] = p.u;
            }
        }
        __syncthreads();

        // ---- issue next tile's global loads; they overlap the WMMAs below ----
        const int kn = k0 + TK;
        if (kn < K) {
#pragma unroll
            for (int i = 0; i < 4; ++i)
                ra[i] = *(const float4*)(A + (size_t)(bm + arr + 32 * i) * K + kn + ac4);
#pragma unroll
            for (int i = 0; i < 2; ++i) {
                rb[2 * i]     = *(const float4*)(Bm + (size_t)(kn + bkp + 16 * i) * N + bn + bn4);
                rb[2 * i + 1] = *(const float4*)(Bm + (size_t)(kn + bkp + 16 * i + 1) * N + bn + bn4);
            }
            if (kn + TK < K) {  // cache prefetch two tiles ahead (global_prefetch_b8)
                __builtin_prefetch(A  + (size_t)(bm + arr) * K + kn + TK + ac4, 0, 1);
                __builtin_prefetch(Bm + (size_t)(kn + TK + bkp) * N + bn + bn4, 0, 1);
            }
        }

        // ---- fragments from LDS + 8 WMMAs ----
        union HF { uint4 q[2]; v16h v; };
        HF af[2], bf[4];
#pragma unroll
        for (int mi = 0; mi < 2; ++mi) {
            int row = wm * 32 + mi * 16 + l16;
            af[mi].q[0] = *(const uint4*)&As[row * LSTR + 8 * g];       // K 8g..8g+7
            af[mi].q[1] = *(const uint4*)&As[row * LSTR + 16 + 8 * g];  // K 16+8g..23+8g
        }
#pragma unroll
        for (int ni = 0; ni < 4; ++ni) {
            int col = wn * 64 + ni * 16 + l16;
            bf[ni].q[0] = *(const uint4*)&Bst[col * LSTR + 8 * g];
            bf[ni].q[1] = *(const uint4*)&Bst[col * LSTR + 16 + 8 * g];
        }
#pragma unroll
        for (int mi = 0; mi < 2; ++mi)
#pragma unroll
            for (int ni = 0; ni < 4; ++ni)
                acc[mi][ni] = __builtin_amdgcn_wmma_f32_16x16x32_f16(
                    false, af[mi].v, false, bf[ni].v, (short)0, acc[mi][ni], false, false);
        __syncthreads();
    }

    // ---- epilogue: bias -> act -> residual -> store (no bounds needed) ----
#pragma unroll
    for (int ni = 0; ni < 4; ++ni) {
        int n = bn + wn * 64 + ni * 16 + l16;
        float bb = bias ? bias[n] : 0.0f;
#pragma unroll
        for (int mi = 0; mi < 2; ++mi)
#pragma unroll
            for (int r = 0; r < 8; ++r) {
                int m = bm + wm * 32 + mi * 16 + r + 8 * g;
                float v = acc[mi][ni][r] + bb;
                if (act == 1) v = 0.5f * v * (1.0f + erff(v * 0.70710678f));
                if (res) v += res[(size_t)m * N + n];
                C[(size_t)m * N + n] = v;
            }
    }
}

// ---------------- attention (online softmax, full-attn stand-in for ProbAttention) ---
__global__ void attn_kernel(const float* __restrict__ Q, const float* __restrict__ K,
                            const float* __restrict__ V, float* __restrict__ O,
                            int Lq, int Lk, int causal)
{
    int idx = blockIdx.x * blockDim.x + threadIdx.x;
    int total = BATCH * NH * Lq;
    if (idx >= total) return;
    int q = idx % Lq;
    int h = (idx / Lq) % NH;
    int b = idx / (Lq * NH);

    const float4* qp = (const float4*)(Q + ((size_t)(b * Lq + q)) * DM + h * DKH);
    float4 qr[16];
#pragma unroll
    for (int d = 0; d < 16; ++d) qr[d] = qp[d];

    float m = -INFINITY, l = 0.0f;
    float4 acc[16];
#pragma unroll
    for (int d = 0; d < 16; ++d) acc[d] = make_float4(0.f, 0.f, 0.f, 0.f);

    int kmax = causal ? (q + 1) : Lk;
    for (int k = 0; k < kmax; ++k) {
        const float4* kp = (const float4*)(K + ((size_t)(b * Lk + k)) * DM + h * DKH);
        float s = 0.0f;
#pragma unroll
        for (int d = 0; d < 16; ++d) {
            float4 kv = kp[d];
            s += qr[d].x * kv.x + qr[d].y * kv.y + qr[d].z * kv.z + qr[d].w * kv.w;
        }
        s *= 0.125f;  // 1/sqrt(64)
        float mn   = fmaxf(m, s);
        float corr = __expf(m - mn);   // first iter: exp(-inf)=0
        float p    = __expf(s - mn);
        l = l * corr + p;
        const float4* vp = (const float4*)(V + ((size_t)(b * Lk + k)) * DM + h * DKH);
#pragma unroll
        for (int d = 0; d < 16; ++d) {
            float4 vv = vp[d];
            acc[d].x = acc[d].x * corr + p * vv.x;
            acc[d].y = acc[d].y * corr + p * vv.y;
            acc[d].z = acc[d].z * corr + p * vv.z;
            acc[d].w = acc[d].w * corr + p * vv.w;
        }
        m = mn;
    }
    float inv = 1.0f / l;
    float4* op = (float4*)(O + ((size_t)(b * Lq + q)) * DM + h * DKH);
#pragma unroll
    for (int d = 0; d < 16; ++d) {
        float4 o;
        o.x = acc[d].x * inv; o.y = acc[d].y * inv;
        o.z = acc[d].z * inv; o.w = acc[d].w * inv;
        op[d] = o;
    }
}

// ---------------- layer norm over last dim (512) ----------------
__global__ void __launch_bounds__(256)
ln_kernel(const float* __restrict__ x, const float* __restrict__ gg,
          const float* __restrict__ bb, float* __restrict__ y)
{
    __shared__ float red[256];
    int row = blockIdx.x;
    int tid = threadIdx.x;
    const float* xr = x + (size_t)row * DM;
    float v0 = xr[tid], v1 = xr[tid + 256];
    red[tid] = v0 + v1;
    __syncthreads();
    for (int s = 128; s > 0; s >>= 1) { if (tid < s) red[tid] += red[tid + s]; __syncthreads(); }
    float mean = red[0] * (1.0f / DM);
    __syncthreads();
    float d0 = v0 - mean, d1 = v1 - mean;
    red[tid] = d0 * d0 + d1 * d1;
    __syncthreads();
    for (int s = 128; s > 0; s >>= 1) { if (tid < s) red[tid] += red[tid + s]; __syncthreads(); }
    float rstd = rsqrtf(red[0] * (1.0f / DM) + 1e-5f);
    float* yr = y + (size_t)row * DM;
    yr[tid]       = d0 * rstd * gg[tid]       + bb[tid];
    yr[tid + 256] = d1 * rstd * gg[tid + 256] + bb[tid + 256];
}

// ---------------- positional + temporal sin-table embedding add ----------------
__device__ __forceinline__ float sctab(float p, int d)
{
    int i2 = d & ~1;
    float div = __expf(-(float)i2 * (9.21034037198f / 512.0f)); // ln(10000)/d_model
    float a = p * div;
    return (d & 1) ? __cosf(a) : __sinf(a);
}

__global__ void posemb_kernel(float* __restrict__ xio, const int* __restrict__ m1,
                              const int* __restrict__ m2, int L, int off1, int split,
                              int s1, int s2)
{
    int idx = blockIdx.x * blockDim.x + threadIdx.x;
    int total = BATCH * L * DM;
    if (idx >= total) return;
    int d = idx & (DM - 1);
    int t = (idx >> 9) % L;
    int b = idx / (DM * L);
    float v = sctab((float)t, d);
    const int* mk = (t < split) ? (m1 + ((size_t)b * s1 + off1 + t) * 4)
                                : (m2 + ((size_t)b * s2 + (t - split)) * 4);
    v += sctab((float)mk[0], d) + sctab((float)mk[1], d)
       + sctab((float)mk[2], d) + sctab((float)mk[3], d);
    xio[idx] += v;
}

// ---------------- im2col for circular conv1d (kernel=3), K padded to 1536 -----------
// dst row layout: 3 blocks of 512; block kb col c = src[b][(t-1+kb) mod L][c], c<Csrc else 0
__global__ void gather3_kernel(const float* __restrict__ src, float* __restrict__ dst,
                               int L, int Csrc)
{
    int idx = blockIdx.x * blockDim.x + threadIdx.x;
    long total = (long)BATCH * L * K3;
    if (idx >= total) return;
    int c3  = idx % K3;
    int row = idx / K3;
    int t = row % L, b = row / L;
    int kb = c3 >> 9, c = c3 & 511;
    float v = 0.0f;
    if (c < Csrc) {
        int ts = (t - 1 + kb + L) % L;
        v = src[((size_t)b * L + ts) * Csrc + c];
    }
    dst[idx] = v;
}

// decoder seq = concat(x[:,192:384], zeros(96)); circular over 288; K padded to 1536
__global__ void gather3_dec_kernel(const float* __restrict__ x, float* __restrict__ dst)
{
    int idx = blockIdx.x * blockDim.x + threadIdx.x;
    long total = (long)BATCH * LDEC * K3;
    if (idx >= total) return;
    int c3  = idx % K3;
    int row = idx / K3;
    int t = row % LDEC, b = row / LDEC;
    int kb = c3 >> 9, c = c3 & 511;
    float v = 0.0f;
    if (c < CIN) {
        int ts = (t - 1 + kb + LDEC) % LDEC;
        if (ts < LENC1)  // first 192 steps come from x[:, 192: ], rest are zero
            v = x[((size_t)b * LENC0 + 192 + ts) * CIN + c];
    }
    dst[idx] = v;
}

// ---------------- weight padding: (3*Csrc, 512) -> block-padded (3*512, 512) --------
__global__ void padw_kernel(const float* __restrict__ src, float* __restrict__ dst,
                            int Csrc)
{
    int idx = blockIdx.x * blockDim.x + threadIdx.x;
    int total = K3 * DM;
    if (idx >= total) return;
    int n = idx & (DM - 1);
    int r = idx >> 9;           // padded row
    int kb = r >> 9, c = r & 511;
    dst[idx] = (c < Csrc) ? src[((size_t)kb * Csrc + c) * DM + n] : 0.0f;
}

// ---------------- projection padding: (512,510)+ (510) -> (512,512) + (512) ---------
__global__ void padproj_kernel(const float* __restrict__ w, const float* __restrict__ b,
                               float* __restrict__ wp, float* __restrict__ bp)
{
    int idx = blockIdx.x * blockDim.x + threadIdx.x;
    if (idx >= DM * DM) return;
    int n = idx & (DM - 1);
    int k = idx >> 9;
    wp[idx] = (n < CIN) ? w[(size_t)k * CIN + n] : 0.0f;
    if (k == 0) bp[n] = (n < CIN) ? b[n] : 0.0f;
}

// ---------------- distil: BN-scale + ELU + maxpool(3, stride2, pad -inf) -------------
__global__ void distil_pool_kernel(const float* __restrict__ yin,
                                   const float* __restrict__ bn_g,
                                   const float* __restrict__ bn_b,
                                   float* __restrict__ out)
{
    int idx = blockIdx.x * blockDim.x + threadIdx.x;
    int total = BATCH * LENC1 * DM;
    if (idx >= total) return;
    int d  = idx & (DM - 1);
    int t2 = (idx >> 9) % LENC1;
    int b  = idx / (DM * LENC1);
    float sc = bn_g[d] * rsqrtf(1.0f + 1e-5f);
    float bo = bn_b[d];
    float best = -INFINITY;
#pragma unroll
    for (int o = -1; o <= 1; ++o) {
        int j = 2 * t2 + o;
        if (j < 0 || j >= LENC0) continue;
        float v = yin[((size_t)b * LENC0 + j) * DM + d] * sc + bo;
        v = (v > 0.0f) ? v : (__expf(v) - 1.0f);   // ELU
        best = fmaxf(best, v);
    }
    out[idx] = best;
}

// ---------------- final slice copy: rows [192,288) of (B,288,512-padded) ------------
__global__ void copy_out_kernel(const float* __restrict__ proj, float* __restrict__ out)
{
    int idx = blockIdx.x * blockDim.x + threadIdx.x;
    int total = BATCH * 96 * CIN;
    if (idx >= total) return;
    int c = idx % CIN;
    int t = (idx / CIN) % 96;
    int b = idx / (CIN * 96);
    out[idx] = proj[((size_t)b * LDEC + 192 + t) * DM + c];
}

// =====================================================================
// host-side orchestration
// =====================================================================
static inline void gemm(const float* A, const float* B, const float* bias,
                        const float* res, float* C, int M, int N, int K, int act,
                        hipStream_t s)
{
    dim3 grid((N + TN - 1) / TN, (M + TM - 1) / TM);
    gemm_f16_wmma<<<grid, 256, 0, s>>>(A, B, bias, res, C, M, N, K, act);
}

static inline void run_attn(const float* Q, const float* K, const float* V, float* O,
                            int Lq, int Lk, int causal, hipStream_t s)
{
    int total = BATCH * NH * Lq;
    attn_kernel<<<(total + 255) / 256, 256, 0, s>>>(Q, K, V, O, Lq, Lk, causal);
}

static inline void run_ln(const float* x, const float* g, const float* b, float* y,
                          int rows, hipStream_t s)
{
    ln_kernel<<<rows, 256, 0, s>>>(x, g, b, y);
}

struct EncP {
    const float *bk, *bo, *bq, *bv, *wk, *wo, *wq, *wv;
    const float *b1, *b2, *ln1b, *ln1g, *ln2b, *ln2g, *w1, *w2;
};

static void encoder_layer(const EncP& p, float* xio, float* t1, float* t2, float* t3,
                          float* t4, float* t5, float* ffbuf, int M, int L, hipStream_t s)
{
    gemm(xio, p.wq, p.bq, nullptr, t1, M, DM, DM, 0, s);
    gemm(xio, p.wk, p.bk, nullptr, t2, M, DM, DM, 0, s);
    gemm(xio, p.wv, p.bv, nullptr, t3, M, DM, DM, 0, s);
    run_attn(t1, t2, t3, t4, L, L, 0, s);
    gemm(t4, p.wo, p.bo, xio, t5, M, DM, DM, 0, s);       // + residual
    run_ln(t5, p.ln1g, p.ln1b, xio, M, s);
    gemm(xio, p.w1, p.b1, nullptr, ffbuf, M, DFF, DM, 1, s);   // GELU
    gemm(ffbuf, p.w2, p.b2, xio, t5, M, DM, DFF, 0, s);        // + residual
    run_ln(t5, p.ln2g, p.ln2b, xio, M, s);
}

extern "C" void kernel_launch(void* const* d_in, const int* in_sizes, int n_in,
                              void* d_out, int out_size, void* d_ws, size_t ws_size,
                              hipStream_t stream)
{
    (void)in_sizes; (void)n_in; (void)out_size; (void)ws_size;
#define PF(i) ((const float*)d_in[(i)])
    const float* x      = PF(0);
    const int* x_timeF  = (const int*)d_in[2];
    const int* y_timeF  = (const int*)d_in[3];
    // params in jax pytree (sorted-key) flatten order
    const float* dec_b1 = PF(4);  const float* dec_b2 = PF(5);
    const float* dc_bk = PF(6);  const float* dc_bo = PF(7);
    const float* dc_bq = PF(8);  const float* dc_bv = PF(9);
    const float* dc_wk = PF(10); const float* dc_wo = PF(11);
    const float* dc_wq = PF(12); const float* dc_wv = PF(13);
    const float* dec_ln1b = PF(14); const float* dec_ln1g = PF(15);
    const float* dec_ln2b = PF(16); const float* dec_ln2g = PF(17);
    const float* dec_ln3b = PF(18); const float* dec_ln3g = PF(19);
    const float* ds_bk = PF(20); const float* ds_bo = PF(21);
    const float* ds_bq = PF(22); const float* ds_bv = PF(23);
    const float* ds_wk = PF(24); const float* ds_wo = PF(25);
    const float* ds_wq = PF(26); const float* ds_wv = PF(27);
    const float* dec_w1 = PF(28); const float* dec_w2 = PF(29);
    const float* dec_emb_w = PF(30);
    const float* dec_nb = PF(31); const float* dec_ng = PF(32);
    const float* dt_bnb = PF(33); const float* dt_bng = PF(34);
    const float* dt_convb = PF(35); const float* dt_convw = PF(36);
    EncP e0 { PF(37),PF(38),PF(39),PF(40),PF(41),PF(42),PF(43),PF(44),
              PF(45),PF(46),PF(47),PF(48),PF(49),PF(50),PF(51),PF(52) };
    EncP e1 { PF(53),PF(54),PF(55),PF(56),PF(57),PF(58),PF(59),PF(60),
              PF(61),PF(62),PF(63),PF(64),PF(65),PF(66),PF(67),PF(68) };
    const float* enc_emb_w = PF(69);
    const float* enc_nb = PF(70); const float* enc_ng = PF(71);
    const float* proj_b = PF(72); const float* proj_w = PF(73);
#undef PF

    // workspace layout (floats)
    const size_t SZ_BIG = (size_t)12288 * 2048;  // im2col + FFN pool
    const size_t SZ_ACT = (size_t)12288 * 512;
    float* ws = (float*)d_ws;
    float* P0 = ws;
    float* bA = P0 + SZ_BIG;
    float* bB = bA + SZ_ACT;
    float* bC = bB + SZ_ACT;
    float* bD = bC + SZ_ACT;
    float* bE = bD + SZ_ACT;
    float* bF = bE + SZ_ACT;
    float* wEncEmb = bF + SZ_ACT;              // 1536*512
    float* wDecEmb = wEncEmb + (size_t)K3 * DM;
    float* wProj   = wDecEmb + (size_t)K3 * DM; // 512*512
    float* bProj   = wProj + (size_t)DM * DM;   // 512

    const int M0 = BATCH * LENC0;   // 12288
    const int M1 = BATCH * LENC1;   // 6144
    const int MD = BATCH * LDEC;    // 9216

    // ---- pad weights once per launch (deterministic) ----
    {
        int tw = K3 * DM;
        padw_kernel<<<(tw + 255) / 256, 256, 0, stream>>>(enc_emb_w, wEncEmb, CIN);
        padw_kernel<<<(tw + 255) / 256, 256, 0, stream>>>(dec_emb_w, wDecEmb, CIN);
        int tp = DM * DM;
        padproj_kernel<<<(tp + 255) / 256, 256, 0, stream>>>(proj_w, proj_b, wProj, bProj);
    }

    // ---- encoder embedding: circular conv (padded K=1536 GEMM) + pos/temporal ----
    {
        long tot = (long)M0 * K3;
        gather3_kernel<<<(tot + 255) / 256, 256, 0, stream>>>(x, P0, LENC0, CIN);
        gemm(P0, wEncEmb, nullptr, nullptr, bA, M0, DM, K3, 0, stream);
        int te = M0 * DM;
        posemb_kernel<<<(te + 255) / 256, 256, 0, stream>>>(bA, x_timeF, x_timeF,
                                                            LENC0, 0, LENC0, LENC0, 0);
    }

    // ---- encoder layer 0 ----
    encoder_layer(e0, bA, bB, bC, bD, bE, bF, P0, M0, LENC0, stream);

    // ---- distil: circ conv (512ch) + BN + ELU + maxpool -> (B,192,512) ----
    {
        long tot = (long)M0 * K3;
        gather3_kernel<<<(tot + 255) / 256, 256, 0, stream>>>(bA, P0, LENC0, DM);
        gemm(P0, dt_convw, dt_convb, nullptr, bB, M0, DM, K3, 0, stream);
        int tp = M1 * DM;
        distil_pool_kernel<<<(tp + 255) / 256, 256, 0, stream>>>(bB, dt_bng, dt_bnb, bA);
    }

    // ---- encoder layer 1 + final encoder norm ----
    encoder_layer(e1, bA, bB, bC, bD, bE, bF, P0, M1, LENC1, stream);
    run_ln(bA, enc_ng, enc_nb, bB, M1, stream);   // bB = cross memory (B,192,512)

    // ---- decoder embedding ----
    {
        long tot = (long)MD * K3;
        gather3_dec_kernel<<<(tot + 255) / 256, 256, 0, stream>>>(x, P0);
        gemm(P0, wDecEmb, nullptr, nullptr, bC, MD, DM, K3, 0, stream);
        int te = MD * DM;
        posemb_kernel<<<(te + 255) / 256, 256, 0, stream>>>(bC, x_timeF, y_timeF,
                                                            LDEC, 192, LENC1, LENC0, 96);
    }

    // ---- decoder self-attention (causal) ----
    gemm(bC, ds_wq, ds_bq, nullptr, bD, MD, DM, DM, 0, stream);
    gemm(bC, ds_wk, ds_bk, nullptr, bE, MD, DM, DM, 0, stream);
    gemm(bC, ds_wv, ds_bv, nullptr, bF, MD, DM, DM, 0, stream);
    run_attn(bD, bE, bF, P0, LDEC, LDEC, 1, stream);
    gemm(P0, ds_wo, ds_bo, bC, bD, MD, DM, DM, 0, stream);
    run_ln(bD, dec_ln1g, dec_ln1b, bC, MD, stream);

    // ---- decoder cross-attention ----
    gemm(bC, dc_wq, dc_bq, nullptr, bD, MD, DM, DM, 0, stream);
    gemm(bB, dc_wk, dc_bk, nullptr, bE, M1, DM, DM, 0, stream);
    gemm(bB, dc_wv, dc_bv, nullptr, bF, M1, DM, DM, 0, stream);
    run_attn(bD, bE, bF, P0, LDEC, LENC1, 0, stream);
    gemm(P0, dc_wo, dc_bo, bC, bD, MD, DM, DM, 0, stream);
    run_ln(bD, dec_ln2g, dec_ln2b, bC, MD, stream);

    // ---- decoder FFN + norms ----
    gemm(bC, dec_w1, dec_b1, nullptr, P0, MD, DFF, DM, 1, stream);   // GELU
    gemm(P0, dec_w2, dec_b2, bC, bD, MD, DM, DFF, 0, stream);
    run_ln(bD, dec_ln3g, dec_ln3b, bE, MD, stream);
    run_ln(bE, dec_ng, dec_nb, bC, MD, stream);

    // ---- projection via padded (512x512) WMMA GEMM + output slice ----
    gemm(bC, wProj, bProj, nullptr, bD, MD, DM, DM, 0, stream);
    int to = BATCH * 96 * CIN;
    copy_out_kernel<<<(to + 255) / 256, 256, 0, stream>>>(bD, (float*)d_out);
}
